// GRU_batchfirst_33895881900054
// MI455X (gfx1250) — matrix operand order repaired and checked by reading, not compiled, not run on hardware
//
#include <hip/hip_runtime.h>

// GRU persistent-scan kernel for gfx1250 (MI455X).
// B=1024,S=1024,I=32,H=128. 64 WGs x 256 threads; each WG owns a 16-row batch
// tile for the whole sequence. Weights resident in VGPRs as bf16 WMMA B-frags,
// hidden state double-buffered in LDS as bf16, f32 accumulation via
// v_wmma_f32_16x16x32_bf16. One barrier per timestep.
//
// Refinements:
//  - software-pipelined x prefetch (global_load ... TH_LOAD_NT issued right
//    after the barrier, consumed reg->LDS next step).
//  - LDS h_{t+1} write-back before the streamed hidden_seq stores.
//  - non-temporal load/store for the 0.64 GB write-once/read-once streams.
//  - both gates via native v_tanh_f32: sigmoid(v) = 0.5*tanh(v/2) + 0.5,
//    replacing the exp/add/rcp chain (2 trans + 1 valu -> 1 trans + 1 fma)
//    on the recurrence-critical elementwise tail.

typedef __attribute__((ext_vector_type(16))) __bf16 v16bf;
typedef __attribute__((ext_vector_type(8)))  float  v8f;

#define B_ 1024
#define S_ 1024
#define I_ 32
#define H_ 128
#define TB 16      // batch rows per workgroup (WMMA M)
#define HPAD 136   // lds_h row stride in elements (272B -> conflict-free)
#define XPAD 40    // lds_x row stride in elements (80B)

static __device__ __forceinline__ float fast_tanh(float v) {
#if __has_builtin(__builtin_amdgcn_tanhf)
  return __builtin_amdgcn_tanhf(v);
#else
  // tanh(v) = 1 - 2*e/(1+e), e = exp(-2v)
  float e = __builtin_amdgcn_exp2f(-2.88539008177793f * v);
  return 1.0f - 2.0f * e * __builtin_amdgcn_rcpf(1.0f + e);
#endif
}

static __device__ __forceinline__ float fast_sigmoid(float v) {
#if __has_builtin(__builtin_amdgcn_tanhf)
  // sigmoid(v) = 0.5*tanh(v/2) + 0.5  (single native v_tanh_f32 + fma)
  return __builtin_fmaf(0.5f, __builtin_amdgcn_tanhf(0.5f * v), 0.5f);
#else
  // sigmoid(v) = 1 / (1 + exp2(-v*log2(e)))
  float e = __builtin_amdgcn_exp2f(-1.44269504088896f * v);
  return __builtin_amdgcn_rcpf(1.0f + e);
#endif
}

__global__ __launch_bounds__(256, 1)
void gru_persistent(const float* __restrict__ x,
                    const float* __restrict__ Wz, const float* __restrict__ Uz,
                    const float* __restrict__ bz,
                    const float* __restrict__ Wh, const float* __restrict__ Uh,
                    const float* __restrict__ bh,
                    float* __restrict__ out)
{
  __shared__ __attribute__((aligned(16))) __bf16 lds_h[2][TB * HPAD];
  __shared__ __attribute__((aligned(16))) __bf16 lds_x[2][TB * XPAD];

  const int tid  = threadIdx.x;
  const int wv   = tid >> 5;          // wave 0..7, owns H columns [16*wv, 16*wv+16)
  const int lane = tid & 31;
  const int lrow = lane & 15;
  const bool lo  = lane < 16;
  const int hi8  = lo ? 0 : 8;        // C/D layout: lanes>=16 hold rows M+8
  const int col  = wv * 16 + lrow;    // this lane's H column (B/C/D N index)
  const int b0   = blockIdx.x * TB;   // batch-row base of this WG

  // per-thread x staging slots: element e = tid + rep*256 of the 16x32 tile
  const int xrow0 = tid >> 5,          xi0 = tid & 31;          // rep 0
  const int xrow1 = (tid + 256) >> 5,  xi1 = tid & 31;          // rep 1

  // ---- one-time: load weight B-fragments as bf16 into registers ----
  // B 32x16 frags in contiguous k order (as A below): lanes 0-15 hold K=0..15 of column N=lane,
  // lanes 16-31 hold K=16..31 of column N=lane-16, 2 values packed per VGPR.
  v16bf fUz[4], fUh[4];
#pragma unroll
  for (int kb = 0; kb < 4; ++kb) {
#pragma unroll
    for (int j = 0; j < 16; ++j) {
      int k = kb * 32 + (lo ? 0 : 16) + j;
      fUz[kb][j] = (__bf16)Uz[k * H_ + col];
      fUh[kb][j] = (__bf16)Uh[k * H_ + col];
    }
  }
  v16bf fWz, fWh;
#pragma unroll
  for (int j = 0; j < 16; ++j) {
    int k = (lo ? 0 : 16) + j;
    fWz[j] = (__bf16)Wz[k * H_ + col];
    fWh[j] = (__bf16)Wh[k * H_ + col];
  }
  const float biasz = bz[col];
  const float biash = bh[col];
  v8f vbz, vbh;
#pragma unroll
  for (int r = 0; r < 8; ++r) { vbz[r] = biasz; vbh[r] = biash; }

  // h_0 = 0
  for (int e = tid; e < TB * HPAD; e += 256) lds_h[0][e] = (__bf16)0.0f;
  float hold[8];
#pragma unroll
  for (int r = 0; r < 8; ++r) hold[r] = 0.0f;

  float* __restrict__ outseq  = out;                         // [B,S,H]
  float* __restrict__ outlast = out + (size_t)B_ * S_ * H_;  // [1,B,H]

  // ---- prime the x pipeline with t=0 ----
  float xr0 = __builtin_nontemporal_load(
      &x[((size_t)(b0 + xrow0) * S_ + 0) * I_ + xi0]);
  float xr1 = __builtin_nontemporal_load(
      &x[((size_t)(b0 + xrow1) * S_ + 0) * I_ + xi1]);

  for (int t = 0; t < S_; ++t) {
    const int cur = t & 1, nxt = cur ^ 1;

    // ---- commit prefetched x_t registers into lds_x[cur] as bf16 ----
    lds_x[cur][xrow0 * XPAD + xi0] = (__bf16)xr0;
    lds_x[cur][xrow1 * XPAD + xi1] = (__bf16)xr1;
    __syncthreads();  // x_t staged; h_t (written end of prev iter) visible

    // ---- issue x_{t+1} prefetch now; latency hides under WMMAs/gates ----
    if (t + 1 < S_) {
      xr0 = __builtin_nontemporal_load(
          &x[((size_t)(b0 + xrow0) * S_ + (t + 1)) * I_ + xi0]);
      xr1 = __builtin_nontemporal_load(
          &x[((size_t)(b0 + xrow1) * S_ + (t + 1)) * I_ + xi1]);
    }

    // ---- A-fragments (contiguous k order, as B): lane<16 -> row M=lane,
    // K {base..base+7} in u[0], K {base+8..base+15} in u[1], base = lo?0:16.
    union { v16bf v; uint4 u[2]; } aX;
    {
      const __bf16* p = &lds_x[cur][lrow * XPAD + (lo ? 0 : 16)];
      aX.u[0] = *(const uint4*)(p);
      aX.u[1] = *(const uint4*)(p + 8);
    }
    union { v16bf v; uint4 u[2]; } aH[4];
#pragma unroll
    for (int kb = 0; kb < 4; ++kb) {
      const __bf16* p = &lds_h[cur][lrow * HPAD + kb * 32 + (lo ? 0 : 16)];
      aH[kb].u[0] = *(const uint4*)(p);
      aH[kb].u[1] = *(const uint4*)(p + 8);
    }

    // ---- gates: acc = bias + x_t@W + h_t@U (f32 accumulate) ----
    v8f accz = vbz, acch = vbh;
    accz = __builtin_amdgcn_wmma_f32_16x16x32_bf16(false, aX.v, false, fWz,
                                                   (short)0, accz, false, false);
    acch = __builtin_amdgcn_wmma_f32_16x16x32_bf16(false, aX.v, false, fWh,
                                                   (short)0, acch, false, false);
#pragma unroll
    for (int kb = 0; kb < 4; ++kb) {
      accz = __builtin_amdgcn_wmma_f32_16x16x32_bf16(false, aH[kb].v, false,
                                                     fUz[kb], (short)0, accz,
                                                     false, false);
      acch = __builtin_amdgcn_wmma_f32_16x16x32_bf16(false, aH[kb].v, false,
                                                     fUh[kb], (short)0, acch,
                                                     false, false);
    }

    // ---- elementwise update; recurrence-critical LDS write-back first ----
#pragma unroll
    for (int r = 0; r < 8; ++r) {
      float z  = fast_sigmoid(accz[r]);
      float th = fast_tanh(acch[r]);
      float hn = __builtin_fmaf(z, th - hold[r], hold[r]);  // (1-z)*h + z*h~
      hold[r] = hn;
      lds_h[nxt][(r + hi8) * HPAD + col] = (__bf16)hn;         // h_{t+1}
    }

    // ---- streamed output (off the critical path, non-temporal) ----
    float* po = outseq + ((size_t)(b0 + hi8) * S_ + t) * H_ + col;
#pragma unroll
    for (int r = 0; r < 8; ++r)
      __builtin_nontemporal_store(hold[r], po + (size_t)r * (S_ * H_));

    if (t == S_ - 1) {
#pragma unroll
      for (int r = 0; r < 8; ++r)
        outlast[(b0 + hi8 + r) * H_ + col] = hold[r];          // h_last
    }
  }
}

extern "C" void kernel_launch(void* const* d_in, const int* in_sizes, int n_in,
                              void* d_out, int out_size, void* d_ws, size_t ws_size,
                              hipStream_t stream) {
  // setup_inputs order: x, W_r, U_r, b_r, W_z, U_z, b_z, W_h, U_h, b_h
  // (r-gate inputs unused — reference discards r)
  const float* x  = (const float*)d_in[0];
  const float* Wz = (const float*)d_in[4];
  const float* Uz = (const float*)d_in[5];
  const float* bz = (const float*)d_in[6];
  const float* Wh = (const float*)d_in[7];
  const float* Uh = (const float*)d_in[8];
  const float* bh = (const float*)d_in[9];

  gru_persistent<<<B_ / TB, 256, 0, stream>>>(x, Wz, Uz, bz, Wh, Uh, bh,
                                              (float*)d_out);
}
